// SparseGCNBlock_18442589569181
// MI455X (gfx1250) — matrix operand order repaired and checked by reading, not compiled, not run on hardware
//
#include <hip/hip_runtime.h>
#include <hip/hip_bf16.h>

typedef __attribute__((ext_vector_type(16))) _Float16 v16h;
typedef __attribute__((ext_vector_type(8)))  _Float16 v8h;
typedef __attribute__((ext_vector_type(8)))  float    v8f;

#define IN_C      128
#define OUT_C     256
#define NUM_EDGES 32
#define LN_EPS    1e-5f
#define WSTRIDE   136   // LDS stride in halves: 272B rows -> 68-bank stride, conflict-free

union HFrag { v16h v; v8h q[2]; _Float16 h[16]; };

// ---------------------------------------------------------------- degrees
__global__ void init_deg_kernel(float* __restrict__ deg, int n) {
    int i = blockIdx.x * blockDim.x + threadIdx.x;
    if (i < n) deg[i] = 1.0f;   // implicit self-loop weight 1
}

__global__ void accum_deg_kernel(const long long* __restrict__ ei,
                                 const float* __restrict__ ew,
                                 float* __restrict__ deg, int e) {
    int i = blockIdx.x * blockDim.x + threadIdx.x;
    if (i < e) {
        int col = (int)ei[(long long)e + i];
        __hip_atomic_fetch_add(&deg[col], ew[i & (NUM_EDGES - 1)],
                               __ATOMIC_RELAXED, __HIP_MEMORY_SCOPE_AGENT);
    }
}

// ---------------------------------------------------------------- fused dual GEMM (WMMA f16)
// xw = x @ W_gcn          (stored raw for edge gather)
// h  = xw * (1/deg) + (x @ W_res) + b_gcn + b_res   (atomics later add edge agg)
__launch_bounds__(256, 1)
__global__ void gemm_kernel(const float* __restrict__ x,
                            const float* __restrict__ Wg,
                            const float* __restrict__ Wr,
                            const float* __restrict__ bg,
                            const float* __restrict__ br,
                            const float* __restrict__ deg,
                            float* __restrict__ xw,
                            float* __restrict__ h,
                            int n) {
    __shared__ __align__(32) _Float16 sW[512 * WSTRIDE]; // [col 0..255 gcn | 256..511 res][K]
    __shared__ __align__(32) _Float16 sX[64 * WSTRIDE];  // [row][K]
    __shared__ float sInv[64];

    const int tid  = threadIdx.x;
    const int row0 = blockIdx.x * 64;

    // Stage weights transposed to [col][K] as f16 (hits in L2 on re-reads).
    for (int j = 0; j < 128; ++j) {                  // 128*256 elems / 256 thr
        int idx = tid + j * 256;
        int k = idx >> 8, c = idx & 255;
        sW[c * WSTRIDE + k]         = (_Float16)Wg[idx];
        sW[(256 + c) * WSTRIDE + k] = (_Float16)Wr[idx];
    }
    // Stage x tile [64][128] as f16.
    for (int j = 0; j < 32; ++j) {                   // 64*128 / 256
        int idx = tid + j * 256;
        int r = idx >> 7, c = idx & 127;
        int g = row0 + r;
        float v = (g < n) ? x[(long long)g * IN_C + c] : 0.0f;
        sX[r * WSTRIDE + c] = (_Float16)v;
    }
    if (tid < 64) {
        int g = row0 + tid;
        sInv[tid] = (g < n) ? (1.0f / deg[g]) : 0.0f;   // dinv*dinv == 1/deg
    }
    __syncthreads();

    const int lane  = tid & 31;
    const int wv    = tid >> 5;       // 8 waves
    const int rtile = wv & 3;         // 4 row tiles of 16
    const int cgrp  = wv >> 2;        // 2 col groups of 8 tiles
    const int lr    = lane & 15;

    // A fragments: contiguous k order (same as B) — lanes 0-15 hold K 0..15, lanes 16-31 hold K 16..31
    HFrag A[4];
    {
        int r    = rtile * 16 + lr;
        int koff = (lane < 16) ? 0 : 16;
#pragma unroll
        for (int kb = 0; kb < 4; ++kb) {
            A[kb].q[0] = *(const v8h*)&sX[r * WSTRIDE + kb * 32 + koff];
            A[kb].q[1] = *(const v8h*)&sX[r * WSTRIDE + kb * 32 + koff + 8];
        }
    }

    const int koffB = (lane < 16) ? 0 : 16;  // B: lanes 0-15 K 0..15, lanes 16-31 K 16..31
    for (int np = 0; np < 8; ++np) {
        int nt  = cgrp * 8 + np;             // 16-wide col tile, 0..15
        int col = nt * 16 + lr;              // output column 0..255

        v8f cg = {}; v8f cr = {};
#pragma unroll
        for (int kb = 0; kb < 4; ++kb) {
            HFrag B;
            B.q[0] = *(const v8h*)&sW[col * WSTRIDE + kb * 32 + koffB];
            B.q[1] = *(const v8h*)&sW[col * WSTRIDE + kb * 32 + koffB + 8];
            cg = __builtin_amdgcn_wmma_f32_16x16x32_f16(false, A[kb].v, false, B.v,
                                                        (short)0, cg, false, false);
        }
#pragma unroll
        for (int kb = 0; kb < 4; ++kb) {
            HFrag B;
            B.q[0] = *(const v8h*)&sW[(256 + col) * WSTRIDE + kb * 32 + koffB];
            B.q[1] = *(const v8h*)&sW[(256 + col) * WSTRIDE + kb * 32 + koffB + 8];
            cr = __builtin_amdgcn_wmma_f32_16x16x32_f16(false, A[kb].v, false, B.v,
                                                        (short)0, cr, false, false);
        }

        float bias = bg[col] + br[col];
#pragma unroll
        for (int v = 0; v < 8; ++v) {
            int M = rtile * 16 + ((lane < 16) ? v : v + 8);
            int g = row0 + M;
            if (g < n) {
                long long o = (long long)g * OUT_C + col;
                float gv = cg[v];
                xw[o] = gv;                                  // raw xw for edge gather
                h[o]  = gv * sInv[M] + cr[v] + bias;         // self-loop + residual + biases
            }
        }
    }
}

// ---------------------------------------------------------------- edge scatter (64 lanes/edge)
__global__ void edge_kernel(const long long* __restrict__ ei,
                            const float* __restrict__ ew,
                            const float* __restrict__ deg,
                            const float* __restrict__ xw,
                            float* __restrict__ h,
                            int e) {
    int t   = blockIdx.x * blockDim.x + threadIdx.x;
    int eid = t >> 6;
    int c   = (t & 63) * 4;
    if (eid >= e) return;
    int row = (int)ei[eid];
    int col = (int)ei[(long long)e + eid];
    float norm = rsqrtf(deg[row]) * ew[eid & (NUM_EDGES - 1)] * rsqrtf(deg[col]);
    const float4 v = *(const float4*)&xw[(long long)row * OUT_C + c];
    long long o = (long long)col * OUT_C + c;
    __hip_atomic_fetch_add(&h[o + 0], norm * v.x, __ATOMIC_RELAXED, __HIP_MEMORY_SCOPE_AGENT);
    __hip_atomic_fetch_add(&h[o + 1], norm * v.y, __ATOMIC_RELAXED, __HIP_MEMORY_SCOPE_AGENT);
    __hip_atomic_fetch_add(&h[o + 2], norm * v.z, __ATOMIC_RELAXED, __HIP_MEMORY_SCOPE_AGENT);
    __hip_atomic_fetch_add(&h[o + 3], norm * v.w, __ATOMIC_RELAXED, __HIP_MEMORY_SCOPE_AGENT);
}

// ---------------------------------------------------------------- LayerNorm + affine + ReLU
__global__ void ln_kernel(const float* __restrict__ h,
                          const float* __restrict__ gamma,
                          const float* __restrict__ beta,
                          float* __restrict__ out, int n) {
    int node = blockIdx.x * 8 + (threadIdx.x >> 5);
    int lane = threadIdx.x & 31;
    if (node >= n) return;

    const float4* hp = (const float4*)(h + (long long)node * OUT_C);
    float4 a = hp[lane * 2 + 0];
    float4 b = hp[lane * 2 + 1];

    float s = a.x + a.y + a.z + a.w + b.x + b.y + b.z + b.w;
    float q = a.x*a.x + a.y*a.y + a.z*a.z + a.w*a.w
            + b.x*b.x + b.y*b.y + b.z*b.z + b.w*b.w;
    for (int off = 16; off > 0; off >>= 1) {
        s += __shfl_xor(s, off);
        q += __shfl_xor(q, off);
    }
    float mu   = s * (1.0f / OUT_C);
    float var  = q * (1.0f / OUT_C) - mu * mu;
    float rstd = rsqrtf(var + LN_EPS);

    const float4* gp = (const float4*)gamma;
    const float4* bp = (const float4*)beta;
    float4 g0 = gp[lane * 2], g1 = gp[lane * 2 + 1];
    float4 e0 = bp[lane * 2], e1 = bp[lane * 2 + 1];

    float4 o0, o1;
    o0.x = fmaxf((a.x - mu) * rstd * g0.x + e0.x, 0.0f);
    o0.y = fmaxf((a.y - mu) * rstd * g0.y + e0.y, 0.0f);
    o0.z = fmaxf((a.z - mu) * rstd * g0.z + e0.z, 0.0f);
    o0.w = fmaxf((a.w - mu) * rstd * g0.w + e0.w, 0.0f);
    o1.x = fmaxf((b.x - mu) * rstd * g1.x + e1.x, 0.0f);
    o1.y = fmaxf((b.y - mu) * rstd * g1.y + e1.y, 0.0f);
    o1.z = fmaxf((b.z - mu) * rstd * g1.z + e1.z, 0.0f);
    o1.w = fmaxf((b.w - mu) * rstd * g1.w + e1.w, 0.0f);

    float4* op = (float4*)(out + (long long)node * OUT_C);
    op[lane * 2 + 0] = o0;
    op[lane * 2 + 1] = o1;
}

// ---------------------------------------------------------------- launcher
extern "C" void kernel_launch(void* const* d_in, const int* in_sizes, int n_in,
                              void* d_out, int out_size, void* d_ws, size_t ws_size,
                              hipStream_t stream) {
    const float*     x     = (const float*)d_in[0];
    const float*     ew    = (const float*)d_in[1];
    const float*     Wg    = (const float*)d_in[2];
    const float*     bg    = (const float*)d_in[3];
    const float*     Wr    = (const float*)d_in[4];
    const float*     br    = (const float*)d_in[5];
    const float*     gamma = (const float*)d_in[6];
    const float*     beta  = (const float*)d_in[7];
    const long long* ei    = (const long long*)d_in[8];

    const int n = in_sizes[0] / IN_C;   // 500000
    const int e = in_sizes[8] / 2;      // 800000

    float* out = (float*)d_out;
    float* h   = (float*)d_ws;                    // [n, 256] accumulation buffer
    float* deg = h + (size_t)n * OUT_C;           // [n]
    float* xw  = out;                             // reuse d_out as xw scratch (overwritten by ln_kernel)

    init_deg_kernel <<<(n + 255) / 256, 256, 0, stream>>>(deg, n);
    accum_deg_kernel<<<(e + 255) / 256, 256, 0, stream>>>(ei, ew, deg, e);
    gemm_kernel     <<<(n + 63) / 64, 256, 0, stream>>>(x, Wg, Wr, bg, br, deg, xw, h, n);
    edge_kernel     <<<(e * 64 + 255) / 256, 256, 0, stream>>>(ei, ew, deg, xw, h, e);
    ln_kernel       <<<(n + 7) / 8, 256, 0, stream>>>(h, gamma, beta, out, n);
}
